// StochasticNeuralODE_39754217292315
// MI455X (gfx1250) — matrix operand order repaired and checked by reading, not compiled, not run on hardware
//
#include <hip/hip_runtime.h>
#include <hip/hip_bf16.h>

// ---------------------------------------------------------------------------
// CDNA5 / gfx1250 Stochastic Neural ODE integrator.
// All GEMMs via v_wmma_f32_16x16x32_bf16; weights resident in LDS (bf16,
// B-transposed, layer-1 biases folded into a constant-one K column);
// per-block time loop with state in registers; branch-free fast tanh;
// quad-tile accumulator groups + B prefetch for WMMA ILP; compiler-only
// LDS fences (wave-private buffers + in-order DS pipe need no dscnt drain).
// ---------------------------------------------------------------------------

typedef __bf16  bf16_t;
typedef bf16_t  v16bf __attribute__((ext_vector_type(16)));
typedef float   v8f   __attribute__((ext_vector_type(8)));

namespace {

constexpr int kB     = 32768;
constexpr int kState = 16;
constexpr int kNC    = 12;
constexpr int kND    = 16;
constexpr int kNT    = 101;
constexpr int kSteps = kNT - 1;

constexpr int kWaves        = 8;
constexpr int kRowsPerWave  = 16;
constexpr int kRowsPerBlock = kWaves * kRowsPerWave;   // 128
constexpr int kBlocks       = kB / kRowsPerBlock;      // 256
constexpr int kThreads      = kWaves * 32;             // 256

// ---- LDS layout (bytes) ----------------------------------------------------
constexpr int OFF_WLIN1 = 0;                       // [256][32]  bf16 = 16384
constexpr int OFF_WNL1  = OFF_WLIN1 + 256*32*2;    // [256][32]  bf16 = 16384
constexpr int OFF_WNL2  = OFF_WNL1  + 256*32*2;    // [256][256] bf16 = 131072
constexpr int OFF_WNL3  = OFF_WNL2  + 256*256*2;   // [16][256]  bf16 = 8192
constexpr int OFF_WLIN2 = OFF_WNL3  + 16*256*2;    // [16][256]  bf16 = 8192
constexpr int OFF_WEN1  = OFF_WLIN2 + 16*256*2;    // [128][32]  bf16 = 8192
constexpr int OFF_WEN2  = OFF_WEN1  + 128*32*2;    // [16][128]  bf16 = 4096
constexpr int OFF_BASE  = OFF_WEN2  + 16*128*2;    // [100][16]  f32  = 6400
constexpr int OFF_BNL2  = OFF_BASE  + kSteps*16*4; // [256]      f32  = 1024
constexpr int OFF_PW    = OFF_BNL2  + 256*4;       // per-wave scratch

// Per-wave private scratch: xtA[16][32]bf16, featsA[16][32]bf16,
// hA[16][256]bf16, bounce[16][64]bf16
constexpr int PW_XT     = 0;
constexpr int PW_FEATS  = 1024;
constexpr int PW_HA     = 2048;
constexpr int PW_BOUNCE = 10240;
constexpr int PW_SIZE   = 12288;

constexpr int kSmemBytes = OFF_PW + kWaves * PW_SIZE;  // 298240 < 320 KiB

struct Params {
  const float* x0;     const float* t;
  const float* lin_w1; const float* lin_b1; const float* lin_w2; const float* lin_b2;
  const float* nl_w1;  const float* nl_b1;  const float* nl_w2;  const float* nl_b2;
  const float* nl_w3;  const float* nl_b3;
  const float* en_w1;  const float* en_b1;  const float* en_w2;  const float* en_b2;
  const float* ne_w1;  const float* ne_b1;  const float* ne_w2;  const float* ne_b2;
  const float* ne_w3;  const float* ne_b3;  const float* amp;
  float* out;
};

union FragBF { v16bf v; uint4 q[2]; };
union F8     { v8f   v; float f[8]; };
union BFBits { bf16_t b; unsigned short s; };

__device__ __forceinline__ unsigned short f2bf(float f) {
  BFBits u; u.b = (bf16_t)f;               // native cvt (v_fma_mix*_bf16 fusable)
  return u.s;
}
__device__ __forceinline__ float bf2f(unsigned short h) {
  return __uint_as_float(((unsigned)h) << 16);
}

// Branch-free tanh: 1 - 2/(exp(2x)+1) via v_exp_f32 + v_rcp_f32.
__device__ __forceinline__ float tanh_fast(float x) {
  float e = __builtin_amdgcn_exp2f(x * 2.8853900817779268f);   // exp(2x)
  return __builtin_fmaf(-2.0f, __builtin_amdgcn_rcpf(e + 1.0f), 1.0f);
}

// Compiler-only ordering fence. Wave-private LDS tiles + the in-order DS
// pipe (ISA 7.3: "LDS: in order") make a hardware dscnt drain unnecessary;
// we only need to stop the compiler from reordering LDS ops across it.
__device__ __forceinline__ void lds_fence() {
  asm volatile("" ::: "memory");
}

// A-fragment (16xK row-major bf16 tile in LDS), K-step s (32 elems).
__device__ __forceinline__ FragBF load_a(const char* lds, int ldK, int s, int lane) {
  int m = lane & 15, kh = lane >> 4;
  FragBF f;
  f.q[0] = *(const uint4*)(lds + (m * ldK + s * 32 + kh * 8) * 2);
  f.q[1] = *(const uint4*)(lds + (m * ldK + s * 32 + 16 + kh * 8) * 2);
  return f;
}

// B-fragment from transposed weight [N][Kpad] bf16; N-tile j, K-step s.
__device__ __forceinline__ FragBF load_b(const char* w, int ldK, int j, int s, int lane) {
  int n = lane & 15, kh = lane >> 4;
  const char* p = w + ((j * 16 + n) * ldK + s * 32 + kh * 8) * 2;
  FragBF f;
  f.q[0] = *(const uint4*)p;
  f.q[1] = *(const uint4*)(p + 32);
  return f;
}

__device__ __forceinline__ v8f wmma_bf16(FragBF a, FragBF b, v8f c) {
  return __builtin_amdgcn_wmma_f32_16x16x32_bf16(false, a.v, false, b.v,
                                                 (short)0, c, false, false);
}

// tanh(D [+ bias]) -> bf16 into a row-major LDS tile (D layout: VGPR r holds
// row m = r + 8*kh, column n = lane&15 within the tile).
__device__ __forceinline__ void stash_tanh0(char* dst, int ldK, int colBase,
                                            v8f d, int lane) {
  F8 u; u.v = d;
  int n = lane & 15, kh = lane >> 4;
#pragma unroll
  for (int r = 0; r < 8; ++r) {
    int m = r + 8 * kh;
    *(unsigned short*)(dst + (m * ldK + colBase + n) * 2) =
        f2bf(tanh_fast(u.f[r]));
  }
}
__device__ __forceinline__ void stash_tanhb(char* dst, int ldK, int colBase,
                                            v8f d, float bias, int lane) {
  F8 u; u.v = d;
  int n = lane & 15, kh = lane >> 4;
#pragma unroll
  for (int r = 0; r < 8; ++r) {
    int m = r + 8 * kh;
    *(unsigned short*)(dst + (m * ldK + colBase + n) * 2) =
        f2bf(tanh_fast(u.f[r] + bias));
  }
}

// Stage a global f32 [K][N] weight into LDS bf16 transposed [N][Kpad].
__device__ __forceinline__ void stage(const float* g, char* dst,
                                      int K, int N, int Kpad, int tid) {
  for (int idx = tid; idx < K * N; idx += kThreads) {
    int k = idx / N, n = idx - k * N;
    *(unsigned short*)(dst + (n * Kpad + k) * 2) = f2bf(g[idx]);
  }
}
__device__ __forceinline__ void stage_bias_row(const float* b, char* dst,
                                               int N, int Kpad, int kRow, int tid) {
  for (int nn = tid; nn < N; nn += kThreads)
    *(unsigned short*)(dst + (nn * Kpad + kRow) * 2) = f2bf(b[nn]);
}

__device__ __forceinline__ unsigned hash_u32(unsigned x) {
  x ^= x >> 16; x *= 0x7feb352dU;
  x ^= x >> 15; x *= 0x846ca68bU;
  x ^= x >> 16; return x;
}

} // namespace

__global__ __launch_bounds__(kThreads)
void sode_cdna5_kernel(Params P) {
  extern __shared__ char smem[];
  const int tid = threadIdx.x;

  // ---- zero LDS (covers K/N zero-padding of weights and tiles) ----
  for (int i = tid; i < kSmemBytes / 4; i += kThreads)
    ((unsigned*)smem)[i] = 0u;
  __syncthreads();

  // ---- stage all weights bf16 / transposed into LDS ----
  stage(P.lin_w1, smem + OFF_WLIN1,  17, 256,  32, tid);
  stage(P.nl_w1,  smem + OFF_WNL1,   17, 256,  32, tid);
  stage(P.nl_w2,  smem + OFF_WNL2,  256, 256, 256, tid);
  stage(P.nl_w3,  smem + OFF_WNL3,  256,  16, 256, tid);
  stage(P.lin_w2, smem + OFF_WLIN2, 256,  16, 256, tid);
  stage(P.en_w1,  smem + OFF_WEN1,    6, 128,  32, tid);
  stage(P.en_w2,  smem + OFF_WEN2,  128,   2, 128, tid);
  stage_bias_row(P.lin_b1, smem + OFF_WLIN1, 256, 32, 17, tid);
  stage_bias_row(P.nl_b1,  smem + OFF_WNL1,  256, 32, 17, tid);
  stage_bias_row(P.en_b1,  smem + OFF_WEN1,  128, 32,  6, tid);
  {
    float* bnl2 = (float*)(smem + OFF_BNL2);
    for (int i = tid; i < 256; i += kThreads) bnl2[i] = P.nl_b2[i];
  }

  // ---- precompute noise gating base[step][state] (time-only network) ----
  float* baseTab = (float*)(smem + OFF_BASE);
  for (int idx = tid; idx < kSteps * 16; idx += kThreads) {
    int i = idx >> 4, n = idx & 15;
    float tk = P.t[i];
    float p[kND];
#pragma unroll
    for (int j = 0; j < kND; ++j)
      p[j] = tanh_fast(tk * P.ne_w1[j] + P.ne_b1[j]);
    float p2[kND];
#pragma unroll
    for (int j = 0; j < kND; ++j) {
      float s = P.ne_b2[j];
#pragma unroll
      for (int k = 0; k < kND; ++k) s += p[k] * P.ne_w2[k * kND + j];
      p2[j] = tanh_fast(s);
    }
    float s = P.ne_b3[n];
#pragma unroll
    for (int k = 0; k < kND; ++k) s += p2[k] * P.ne_w3[k * kState + n];
    float pat = tanh_fast(s);
    int cyc = (int)fmodf(tk * (float)kNC, (float)kNC);
    if (cyc < 0) cyc = 0;
    if (cyc >= kNC) cyc = kNC - 1;
    baseTab[idx] = P.amp[n * kNC + cyc] * pat;   // NOISE_SCALE == 1
  }
  __syncthreads();

  // ---- per-wave setup ----
  const int wave = tid >> 5;
  const int lane = tid & 31;
  const int n  = lane & 15;
  const int kh = lane >> 4;
  char* xtA    = smem + OFF_PW + wave * PW_SIZE + PW_XT;
  char* featsA = smem + OFF_PW + wave * PW_SIZE + PW_FEATS;
  char* hA     = smem + OFF_PW + wave * PW_SIZE + PW_HA;
  char* bounce = smem + OFF_PW + wave * PW_SIZE + PW_BOUNCE;
  const char* WLIN1 = smem + OFF_WLIN1;
  const char* WNL1  = smem + OFF_WNL1;
  const char* WNL2  = smem + OFF_WNL2;
  const char* WNL3  = smem + OFF_WNL3;
  const char* WLIN2 = smem + OFF_WLIN2;
  const char* WEN1  = smem + OFF_WEN1;
  const char* WEN2  = smem + OFF_WEN2;
  const float* bnl2 = (const float*)(smem + OFF_BNL2);

  // constant-1 columns (xt col 17, feats col 6) -- written once
  if (lane < 16) {
    *(unsigned short*)(xtA    + (lane * 32 + 17) * 2) = 0x3F80;
    *(unsigned short*)(featsA + (lane * 32 +  6) * 2) = 0x3F80;
  }

  const int rowBase = blockIdx.x * kRowsPerBlock + wave * kRowsPerWave;

  // State kept in registers in WMMA C/D layout: f[r] is row (r + 8*kh), col n.
  F8 x;
#pragma unroll
  for (int r = 0; r < 8; ++r) {
    int gb = rowBase + r + 8 * kh;
    x.f[r] = P.x0[gb * kState + n];
    P.out[(gb * kNT + 0) * kState + n] = x.f[r];
  }

  const float dt  = P.t[1] - P.t[0];
  const float lb2 = P.lin_b2[n];
  const float nb3 = P.nl_b3[n];
  const float eb2 = (n < 2) ? P.en_b2[n] : 0.0f;
  const v8f zero8 = {};

  for (int i = 0; i < kSteps; ++i) {
    const float tk = P.t[i];

    // -------- build xt tile (cols 0..15 = x, col 16 = tk, col 17 = 1) ------
#pragma unroll
    for (int r = 0; r < 8; ++r) {
      int m = r + 8 * kh;
      *(unsigned short*)(xtA + (m * 32 + n) * 2) = f2bf(x.f[r]);
    }
    if (n == 0) {
      unsigned short tkb = f2bf(tk);
#pragma unroll
      for (int r = 0; r < 8; ++r) {
        int m = r + 8 * kh;
        *(unsigned short*)(xtA + (m * 32 + 16) * 2) = tkb;
      }
    }
    lds_fence();

    // -------- ENSO feature tile (cols 0..5 live, col 6 = 1) ----------------
    if (lane < 16) {
      int m = lane;
      float T = bf2f(*(const unsigned short*)(xtA + (m * 32 + 0) * 2));
      float H = bf2f(*(const unsigned short*)(xtA + (m * 32 + 1) * 2));
      unsigned short* fp = (unsigned short*)(featsA + m * 32 * 2);
      fp[0] = f2bf(T * T);     fp[1] = f2bf(T * H);
      fp[2] = f2bf(T * T * T); fp[3] = f2bf(T * T * H);
      fp[4] = f2bf(T * H * H); fp[5] = f2bf(tk);
    }
    lds_fence();

    FragBF axt = load_a(xtA, 32, 0, lane);

    // -------- nonlinear path layer 1: h1 = tanh(xt @ [nl_w1;b1]) -> hA -----
#pragma unroll
    for (int j = 0; j < 16; ++j) {
      v8f d = wmma_bf16(axt, load_b(WNL1, 32, j, 0, lane), zero8);
      stash_tanh0(hA, 256, j * 16, d, lane);
    }
    lds_fence();
    FragBF ah[8];
#pragma unroll
    for (int s = 0; s < 8; ++s) ah[s] = load_a(hA, 256, s, lane);

    // -------- layers 2+3 fused: quad-tile groups, B prefetched 1 K-step ----
    v8f nlacc = zero8;
#pragma unroll
    for (int jq = 0; jq < 4; ++jq) {
      const int j0 = 4 * jq;
      v8f d0 = zero8, d1 = zero8, d2 = zero8, d3 = zero8;
      FragBF b0 = load_b(WNL2, 256, j0 + 0, 0, lane);
      FragBF b1 = load_b(WNL2, 256, j0 + 1, 0, lane);
      FragBF b2 = load_b(WNL2, 256, j0 + 2, 0, lane);
      FragBF b3 = load_b(WNL2, 256, j0 + 3, 0, lane);
#pragma unroll
      for (int s = 0; s < 8; ++s) {
        FragBF c0 = b0, c1 = b1, c2 = b2, c3 = b3;
        if (s < 7) {
          c0 = load_b(WNL2, 256, j0 + 0, s + 1, lane);
          c1 = load_b(WNL2, 256, j0 + 1, s + 1, lane);
          c2 = load_b(WNL2, 256, j0 + 2, s + 1, lane);
          c3 = load_b(WNL2, 256, j0 + 3, s + 1, lane);
        }
        d0 = wmma_bf16(ah[s], b0, d0);
        d1 = wmma_bf16(ah[s], b1, d1);
        d2 = wmma_bf16(ah[s], b2, d2);
        d3 = wmma_bf16(ah[s], b3, d3);
        b0 = c0; b1 = c1; b2 = c2; b3 = c3;
      }
      stash_tanhb(bounce, 64,  0, d0, bnl2[(j0 + 0) * 16 + n], lane);
      stash_tanhb(bounce, 64, 16, d1, bnl2[(j0 + 1) * 16 + n], lane);
      stash_tanhb(bounce, 64, 32, d2, bnl2[(j0 + 2) * 16 + n], lane);
      stash_tanhb(bounce, 64, 48, d3, bnl2[(j0 + 3) * 16 + n], lane);
      lds_fence();
      FragBF ab0 = load_a(bounce, 64, 0, lane);
      FragBF ab1 = load_a(bounce, 64, 1, lane);
      nlacc = wmma_bf16(ab0, load_b(WNL3, 256, 0, 2 * jq,     lane), nlacc);
      nlacc = wmma_bf16(ab1, load_b(WNL3, 256, 0, 2 * jq + 1, lane), nlacc);
      lds_fence();
    }

    // -------- linear path: tanh(xt @ [lin_w1;b1]) @ lin_w2 -----------------
    v8f linacc = zero8;
#pragma unroll
    for (int jq = 0; jq < 4; ++jq) {
      const int j0 = 4 * jq;
      v8f d0 = wmma_bf16(axt, load_b(WLIN1, 32, j0 + 0, 0, lane), zero8);
      v8f d1 = wmma_bf16(axt, load_b(WLIN1, 32, j0 + 1, 0, lane), zero8);
      v8f d2 = wmma_bf16(axt, load_b(WLIN1, 32, j0 + 2, 0, lane), zero8);
      v8f d3 = wmma_bf16(axt, load_b(WLIN1, 32, j0 + 3, 0, lane), zero8);
      stash_tanh0(bounce, 64,  0, d0, lane);
      stash_tanh0(bounce, 64, 16, d1, lane);
      stash_tanh0(bounce, 64, 32, d2, lane);
      stash_tanh0(bounce, 64, 48, d3, lane);
      lds_fence();
      FragBF ab0 = load_a(bounce, 64, 0, lane);
      FragBF ab1 = load_a(bounce, 64, 1, lane);
      linacc = wmma_bf16(ab0, load_b(WLIN2, 256, 0, 2 * jq,     lane), linacc);
      linacc = wmma_bf16(ab1, load_b(WLIN2, 256, 0, 2 * jq + 1, lane), linacc);
      lds_fence();
    }

    // -------- ENSO path: tanh(feats @ [en_w1;b1]) @ en_w2 ------------------
    FragBF afe = load_a(featsA, 32, 0, lane);
    v8f enacc = zero8;
#pragma unroll
    for (int jq = 0; jq < 2; ++jq) {
      const int j0 = 4 * jq;
      v8f d0 = wmma_bf16(afe, load_b(WEN1, 32, j0 + 0, 0, lane), zero8);
      v8f d1 = wmma_bf16(afe, load_b(WEN1, 32, j0 + 1, 0, lane), zero8);
      v8f d2 = wmma_bf16(afe, load_b(WEN1, 32, j0 + 2, 0, lane), zero8);
      v8f d3 = wmma_bf16(afe, load_b(WEN1, 32, j0 + 3, 0, lane), zero8);
      stash_tanh0(bounce, 64,  0, d0, lane);
      stash_tanh0(bounce, 64, 16, d1, lane);
      stash_tanh0(bounce, 64, 32, d2, lane);
      stash_tanh0(bounce, 64, 48, d3, lane);
      lds_fence();
      FragBF ab0 = load_a(bounce, 64, 0, lane);
      FragBF ab1 = load_a(bounce, 64, 1, lane);
      enacc = wmma_bf16(ab0, load_b(WEN2, 128, 0, 2 * jq,     lane), enacc);
      enacc = wmma_bf16(ab1, load_b(WEN2, 128, 0, 2 * jq + 1, lane), enacc);
      lds_fence();
    }

    // -------- Euler update + stochastic forcing + trajectory store ---------
    const float bse = baseTab[i * 16 + n];
    F8 ln; ln.v = linacc;
    F8 nn; nn.v = nlacc;
    F8 en; en.v = enacc;
#pragma unroll
    for (int rp = 0; rp < 4; ++rp) {
      int r0 = 2 * rp, r1 = 2 * rp + 1;
      int gb0 = rowBase + r0 + 8 * kh;
      int gb1 = gb0 + 1;
      unsigned e  = hash_u32(((unsigned)(gb0 * kState + n)) * 2654435761u ^
                             ((unsigned)i) * 0x9E3779B9u);
      unsigned e2 = hash_u32(e ^ 0x68bc21ebu);
      float u1 = ((e >> 8) + 1u) * (1.0f / 16777216.0f);
      float u2 = (e2 >> 8) * (1.0f / 16777216.0f);
      float R  = __builtin_amdgcn_sqrtf(-2.0f * __logf(u1));
      float a  = 6.28318530718f * u2;
      float rn0 = R * __cosf(a);
      float rn1 = R * __sinf(a);

      float ens0 = (n < 2) ? (en.f[r0] + eb2) : 0.0f;
      float ens1 = (n < 2) ? (en.f[r1] + eb2) : 0.0f;
      float vf0 = (ln.f[r0] + lb2) + (nn.f[r0] + nb3) + ens0 + rn0 * bse;
      float vf1 = (ln.f[r1] + lb2) + (nn.f[r1] + nb3) + ens1 + rn1 * bse;
      x.f[r0] += dt * vf0;
      x.f[r1] += dt * vf1;
      P.out[(gb0 * kNT + (i + 1)) * kState + n] = x.f[r0];
      P.out[(gb1 * kNT + (i + 1)) * kState + n] = x.f[r1];
    }
  }
}

extern "C" void kernel_launch(void* const* d_in, const int* in_sizes, int n_in,
                              void* d_out, int out_size, void* d_ws, size_t ws_size,
                              hipStream_t stream) {
  (void)in_sizes; (void)n_in; (void)out_size; (void)d_ws; (void)ws_size;
  Params P;
  P.x0     = (const float*)d_in[0];
  P.t      = (const float*)d_in[1];
  P.lin_w1 = (const float*)d_in[2];
  P.lin_b1 = (const float*)d_in[3];
  P.lin_w2 = (const float*)d_in[4];
  P.lin_b2 = (const float*)d_in[5];
  P.nl_w1  = (const float*)d_in[6];
  P.nl_b1  = (const float*)d_in[7];
  P.nl_w2  = (const float*)d_in[8];
  P.nl_b2  = (const float*)d_in[9];
  P.nl_w3  = (const float*)d_in[10];
  P.nl_b3  = (const float*)d_in[11];
  P.en_w1  = (const float*)d_in[12];
  P.en_b1  = (const float*)d_in[13];
  P.en_w2  = (const float*)d_in[14];
  P.en_b2  = (const float*)d_in[15];
  P.ne_w1  = (const float*)d_in[16];
  P.ne_b1  = (const float*)d_in[17];
  P.ne_w2  = (const float*)d_in[18];
  P.ne_b2  = (const float*)d_in[19];
  P.ne_w3  = (const float*)d_in[20];
  P.ne_b3  = (const float*)d_in[21];
  P.amp    = (const float*)d_in[22];
  P.out    = (float*)d_out;

  sode_cdna5_kernel<<<dim3(kBlocks), dim3(kThreads), kSmemBytes, stream>>>(P);
}
